// HalfInteger2Bit8ColLinear_45990509806219
// MI455X (gfx1250) — compile-verified, matrix operand order checked by reading, not run on hardware
//
#include <hip/hip_runtime.h>
#include <hip/hip_bf16.h>

typedef __attribute__((ext_vector_type(16))) _Float16 v16h;
typedef __attribute__((ext_vector_type(8)))  _Float16 v8h;
typedef __attribute__((ext_vector_type(8)))  float    v8f;
typedef __attribute__((ext_vector_type(4)))  int      v4i;
typedef __attribute__((ext_vector_type(4)))  unsigned v4u;

#define T_DIM 256
#define D_IN  8192   // n * CODESZ
#define M_DIM 8192
#define N_Q   1024   // Qidxs columns
// 1/sqrt(8192)
#define INV_SQRT_D 0.011048543456039806f
#define NUM_SCALE  1024.0f

// ---------------------------------------------------------------------------
// Kernel 1: Xh[row] = fp16( FWHT(input[row] * SU) / (sqrt(8192) * 1024) )
// ---------------------------------------------------------------------------
__global__ void fwht_pre_kernel(const float* __restrict__ x,
                                const float* __restrict__ su,
                                _Float16* __restrict__ xh) {
    __shared__ float buf[D_IN];
    const int row = blockIdx.x;
    const float* xr = x + (size_t)row * D_IN;
    for (int i = threadIdx.x; i < D_IN; i += 256)
        buf[i] = xr[i] * su[i];
    __syncthreads();
    for (int h = 1; h < D_IN; h <<= 1) {
        for (int p = threadIdx.x; p < D_IN / 2; p += 256) {
            int i = ((p & ~(h - 1)) << 1) | (p & (h - 1));
            float a = buf[i], b = buf[i + h];
            buf[i] = a + b;
            buf[i + h] = a - b;
        }
        __syncthreads();
    }
    const float s = INV_SQRT_D / NUM_SCALE;
    _Float16* o = xh + (size_t)row * D_IN;
    for (int i = threadIdx.x; i < D_IN; i += 256)
        o[i] = (_Float16)(buf[i] * s);
}

// ---------------------------------------------------------------------------
// Dequant of one Qidxs entry -> 8 fp16 values: codebook row XOR sign mask,
// both from LDS tables (1 ds_load_b128 each + 4 v_xor).
// ---------------------------------------------------------------------------
__device__ __forceinline__ v8h decode8(int q, const _Float16* grid,
                                       const v4u* signs) {
    unsigned idx = (unsigned)(q + 32768);        // q - IDX_OFFSET
    v4u g = *(const v4u*)(grid + (idx & 255u) * 8);
    v4u s = signs[(idx >> 8) & 255u];
    g ^= s;
    return __builtin_bit_cast(v8h, g);
}

// ---------------------------------------------------------------------------
// Kernel 2: Z(256 x 8192) f32 = Xh(256 x 8192) f16 @ dequant(Qidxs)^T
// Block = 256 threads = 8 waves (exactly 2 waves per SIMD32 on a WGP).
// amdgpu_waves_per_eu(2) lifts the allocator's occupancy-driven VGPR cap to
// what 2 waves/SIMD allows -- this kernel needs ~200 VGPRs (64 acc +
// in-flight A/B fragments) and hides latency via ILP, not occupancy.
// Each wave: 64(T) x 32(m) = 4 T-tiles x 2 m-tiles; each decoded B fragment
// feeds 4 WMMAs; Qidxs quads double-buffered one 32-K step ahead; tile
// strides folded into the 24-bit instruction offsets.
// Grid = (M/256, T/64) = (32, 4).
// ---------------------------------------------------------------------------
#define COMPUTE_STEP(qb, kk)                                                 \
    do {                                                                     \
        v16h a[4];                                                           \
        _Pragma("unroll") for (int tt = 0; tt < 4; ++tt) {                   \
            v8h alo = *(const v8h*)(Xh + (xbase + (kk)) + tt * (16 * D_IN)); \
            v8h ahi = *(const v8h*)(Xh + (xbase + (kk)) + tt * (16 * D_IN)   \
                                    + 16);                                   \
            a[tt] = __builtin_shufflevector(alo, ahi,                        \
                0, 1, 2, 3, 4, 5, 6, 7, 8, 9, 10, 11, 12, 13, 14, 15);       \
        }                                                                    \
        _Pragma("unroll") for (int t = 0; t < 2; ++t) {                      \
            int qlo = hilo ? qb[t][1] : qb[t][0];                            \
            int qhi = hilo ? qb[t][3] : qb[t][2];                            \
            v8h blo = decode8(qlo, gridLds, signLds);                        \
            v8h bhi = decode8(qhi, gridLds, signLds);                        \
            v16h b = __builtin_shufflevector(blo, bhi,                       \
                0, 1, 2, 3, 4, 5, 6, 7, 8, 9, 10, 11, 12, 13, 14, 15);       \
            _Pragma("unroll") for (int tt = 0; tt < 4; ++tt)                 \
                acc[tt][t] = __builtin_amdgcn_wmma_f32_16x16x32_f16(         \
                    false, a[tt], false, b, (short)0, acc[tt][t],            \
                    false, false);                                           \
        }                                                                    \
    } while (0)

#define LOAD_Q(qb, kk)                                                       \
    do {                                                                     \
        _Pragma("unroll") for (int t = 0; t < 2; ++t)                        \
            qb[t] = *(const v4i*)(Qidxs + (qbase + ((kk) >> 3))              \
                                  + t * (16 * N_Q));                         \
    } while (0)

__global__ void __launch_bounds__(256)
__attribute__((amdgpu_waves_per_eu(2)))
gemm_dequant_kernel(const _Float16* __restrict__ Xh,
                    const int* __restrict__ Qidxs,
                    const _Float16* __restrict__ gp, int gpElems,
                    float* __restrict__ Z) {
    __shared__ _Float16 gridLds[256 * 8];   // 4 KB codebook
    __shared__ v4u      signLds[256];       // 4 KB sign-flip masks
    const int tid = threadIdx.x;
    for (int i = tid; i < 256 * 8; i += 256)
        gridLds[i] = (i < gpElems) ? gp[i] : (_Float16)0.0f;
    {
        const unsigned f = (unsigned)tid;   // blockDim.x == 256, covers all
        v4u m;
#pragma unroll
        for (int i = 0; i < 4; ++i)
            m[i] = (((f >> (2 * i)) & 1u) << 15) |
                   (((f >> (2 * i + 1)) & 1u) << 31);
        signLds[tid] = m;
    }
    __syncthreads();

    const int lane   = tid & 31;
    const int wave   = tid >> 5;
    const int lane16 = lane & 15;
    const int hilo   = lane >> 4;       // which half-wave (K sub-block)
    const int off    = hilo << 3;       // 0 or 8

    const int tBase = blockIdx.y * 64;              // 4 T-tiles of 16
    const int mBase = blockIdx.x * 256 + wave * 32; // 2 m-tiles of 16

    // 32-bit element offsets; tile strides become instruction immediates.
    const unsigned xbase = (unsigned)(tBase + lane16) * D_IN + (unsigned)off;
    const unsigned qbase = (unsigned)(mBase + lane16) * N_Q;

    v8f acc[4][2] = {};

    v4i qb0[2], qb1[2];
    LOAD_Q(qb0, 0);

    for (int k0 = 0; k0 < D_IN; k0 += 64) {
        LOAD_Q(qb1, k0 + 32);           // always in range (k0 <= D_IN-64)
        COMPUTE_STEP(qb0, k0);
        if (k0 + 64 < D_IN)
            LOAD_Q(qb0, k0 + 64);
        COMPUTE_STEP(qb1, k0 + 32);
    }

    // C/D layout: VGPR r -> row (r + 8*hilo), col lane16.
#pragma unroll
    for (int tt = 0; tt < 4; ++tt) {
#pragma unroll
        for (int t = 0; t < 2; ++t) {
            const unsigned mcol = (unsigned)(mBase + 16 * t + lane16);
#pragma unroll
            for (int r = 0; r < 8; ++r) {
                const unsigned trow = (unsigned)(tBase + 16 * tt + r + 8 * hilo);
                Z[(size_t)trow * M_DIM + mcol] = acc[tt][t][r];
            }
        }
    }
}

// ---------------------------------------------------------------------------
// Kernel 3: out[row] = FWHT(Z[row]) * (Wscale * 1024 / sqrt(8192)) * SV
// ---------------------------------------------------------------------------
__global__ void fwht_post_kernel(const float* __restrict__ z,
                                 const float* __restrict__ sv,
                                 const float* __restrict__ wscale,
                                 float* __restrict__ out) {
    __shared__ float buf[M_DIM];
    const int row = blockIdx.x;
    const float* zr = z + (size_t)row * M_DIM;
    for (int i = threadIdx.x; i < M_DIM; i += 256)
        buf[i] = zr[i];
    __syncthreads();
    for (int h = 1; h < M_DIM; h <<= 1) {
        for (int p = threadIdx.x; p < M_DIM / 2; p += 256) {
            int i = ((p & ~(h - 1)) << 1) | (p & (h - 1));
            float a = buf[i], b = buf[i + h];
            buf[i] = a + b;
            buf[i + h] = a - b;
        }
        __syncthreads();
    }
    const float s = wscale[0] * NUM_SCALE * INV_SQRT_D;
    float* o = out + (size_t)row * M_DIM;
    for (int i = threadIdx.x; i < M_DIM; i += 256)
        o[i] = buf[i] * s * sv[i];
}

// ---------------------------------------------------------------------------
extern "C" void kernel_launch(void* const* d_in, const int* in_sizes, int n_in,
                              void* d_out, int out_size, void* d_ws, size_t ws_size,
                              hipStream_t stream) {
    const float*    input  = (const float*)d_in[0];     // (256, 8192) f32
    const int*      Qidxs  = (const int*)d_in[1];       // (8192, 1024) i32
    const float*    SU     = (const float*)d_in[2];     // (8192,) f32
    const float*    SV     = (const float*)d_in[3];     // (8192,) f32
    const float*    Wscale = (const float*)d_in[4];     // (1,) f32
    const _Float16* gp     = (const _Float16*)d_in[5];  // (gpN, 8) f16
    const int gpElems      = in_sizes[5];

    _Float16* Xh = (_Float16*)d_ws;                                   // 4 MB
    float*    Z  = (float*)((char*)d_ws +
                            (size_t)T_DIM * D_IN * sizeof(_Float16)); // 8 MB
    float*    out = (float*)d_out;

    fwht_pre_kernel<<<T_DIM, 256, 0, stream>>>(input, SU, Xh);
    gemm_dequant_kernel<<<dim3(M_DIM / 256, T_DIM / 64), 256, 0, stream>>>(
        Xh, Qidxs, gp, gpElems, Z);
    fwht_post_kernel<<<T_DIM, 256, 0, stream>>>(Z, SV, Wscale, out);
}
